// Channel_Attention_23673859735840
// MI455X (gfx1250) — matrix operand order repaired and checked by reading, not compiled, not run on hardware
//
#include <hip/hip_runtime.h>

// ---------- types ----------
typedef __attribute__((ext_vector_type(16))) __bf16 v16bf;
typedef __attribute__((ext_vector_type(8)))  float  v8f;

union Frag {
    v16bf v;
    uint4 q[2];
};

__device__ __forceinline__ unsigned short f2bf(float f) {
    union { float f; unsigned u; } c; c.f = f;
    unsigned u = c.u;
    unsigned r = (u + 0x7FFFu + ((u >> 16) & 1u)) >> 16;   // round-to-nearest-even
    return (unsigned short)r;
}
__device__ __forceinline__ float bf2f(unsigned short h) {
    union { unsigned u; float f; } c; c.u = ((unsigned)h) << 16;
    return c.f;
}

// ---------- fp32 -> bf16 conversion ----------
__global__ __launch_bounds__(256) void f2bf_kernel(const float* __restrict__ src,
                                                   unsigned short* __restrict__ dst,
                                                   unsigned long long n) {
    unsigned long long i = (unsigned long long)blockIdx.x * 256ull + threadIdx.x;
    if (i < n) dst[i] = f2bf(src[i]);
}

// ---------- bf16 GEMM:  out[b, Mrows, 16384] = W[Mrows,256] @ X[b,256,16384] ----------
// One wave owns one 16-wide column strip (b, nt). It preloads the whole 256x16
// bf16 B-strip into 8 fragments (64 VGPRs) and sweeps all M tiles. Per M tile,
// all 8 A fragments are issued as clustered b128 loads BEFORE the 8 WMMAs, so
// the compiler can overlap the (L2-resident) weight stream with the matrix pipe
// using partial s_wait_loadcnt. Next tile's weights are prefetched.
template <int STORE_BF16>
__global__ __launch_bounds__(128) void gemm_bf16_kernel(const unsigned short* __restrict__ Wbf,
                                                        const unsigned short* __restrict__ Xbf,
                                                        void* __restrict__ out,
                                                        int Mtiles) {
    const int lane = threadIdx.x & 31;
    const int wid  = threadIdx.x >> 5;
    const int task = blockIdx.x * 4 + wid;       // 8192 tasks = 8 b * 1024 strips
    const int b    = task >> 10;
    const int nt   = task & 1023;
    const size_t hw0 = (size_t)nt * 16;
    const int m    = lane & 15;
    const int half = lane >> 4;
    const int Mrows = Mtiles * 16;

    // B-strip: lane L holds column n = L&15; element i -> row c = 32*t + i + 8*half + (i>=8 ? 8 : 0).
    Frag B[8];
#pragma unroll
    for (int t = 0; t < 8; ++t) {
#pragma unroll
        for (int i = 0; i < 16; ++i) {
            const int c = 32 * t + i + 8 * half + (i >= 8 ? 8 : 0);
            B[t].v[i] = __builtin_bit_cast(__bf16,
                Xbf[((size_t)b * 256 + (size_t)c) * 16384 + hw0 + m]);
        }
    }

    // Per-lane base into W for this lane's A-fragment rows:
    //   A 16x32 bf16 layout: lane(m,half): elems 0..7 -> K = 8*half+0..7,
    //   elems 8..15 -> K = 16+8*half+0..7  (two contiguous 16B chunks)
    const unsigned short* wbase = Wbf + (size_t)m * 256 + 8 * half;

    for (int mt = 0; mt < Mtiles; ++mt) {
        const unsigned short* wrow = wbase + (size_t)mt * 16 * 256;

        // Stage all A fragments for this tile (16 x b128, clustered).
        Frag a[8];
#pragma unroll
        for (int t = 0; t < 8; ++t) {
            const unsigned short* wp = wrow + 32 * t;
            a[t].q[0] = *(const uint4*)(wp);
            a[t].q[1] = *(const uint4*)(wp + 16);
        }
        // Prefetch next tile's weight rows into cache (global_prefetch_b8).
        if (mt + 1 < Mtiles)
            __builtin_prefetch(wrow + (size_t)16 * 256, 0, 1);

        v8f acc = {0.f, 0.f, 0.f, 0.f, 0.f, 0.f, 0.f, 0.f};
#pragma unroll
        for (int t = 0; t < 8; ++t)
            acc = __builtin_amdgcn_wmma_f32_16x16x32_bf16(
                false, a[t].v, false, B[t].v, (short)0, acc, false, false);

        // C layout: vgpr r -> row = r + 8*half, col = lane&15
#pragma unroll
        for (int r = 0; r < 8; ++r) {
            int row = mt * 16 + r + 8 * half;
            size_t off = ((size_t)b * Mrows + row) * 16384 + hw0 + (lane & 15);
            if (STORE_BF16) ((unsigned short*)out)[off] = f2bf(acc[r]);
            else            ((float*)out)[off]          = acc[r];
        }
    }
}

// ---------- fused depthwise-conv + window channel-attention ----------
// Block = (b, window, head-group of 4); one wave per head.
// q/k live in LDS as 32x32 (columns 16..31 zeroed) so K-padded WMMA fragment
// builds are branch-free; conv halo loads are clamp+mask (no exec branches).
__global__ __launch_bounds__(128) void attn_kernel(const unsigned short* __restrict__ qkv_bf,
                                                   const float* __restrict__ Wdw,
                                                   const float* __restrict__ temperature,
                                                   unsigned short* __restrict__ attnout) {
    // per wave: q 32x32 (1024), k 32x32 (1024), v 32x16 (512), attn 32x32 (1024)
    __shared__ float sh[4 * 3584];
    const int lane = threadIdx.x & 31;
    const int wid  = threadIdx.x >> 5;
    const unsigned bid = blockIdx.x;               // 8 * 1024 * 2 = 16384 blocks
    const int hg = bid & 1;
    const int n  = (bid >> 1) & 1023;
    const int b  = bid >> 11;
    const int h  = hg * 4 + wid;
    const int wy = n >> 5, wx = n & 31;

    float* lq  = sh + wid * 3584;       // stride 32, e in [0,32)
    float* lk  = lq + 1024;             // stride 32
    float* lv  = lk + 1024;             // stride 16
    float* lat = lv + 512;              // 32x32

    const float temp = temperature[h];

    // ---- zero the K-pad columns (e = 16..31) of q and k: lane = row d ----
#pragma unroll
    for (int e = 16; e < 32; ++e) {
        lq[lane * 32 + e] = 0.f;
        lk[lane * 32 + e] = 0.f;
    }

    // ---- depthwise 3x3 conv: lane owns channel d = lane for each of q/k/v ----
    for (int s = 0; s < 3; ++s) {
        const int d  = lane;
        const int ch = s * 256 + h * 32 + d;
        float w[9];
#pragma unroll
        for (int j = 0; j < 9; ++j) w[j] = Wdw[ch * 9 + j];

        const unsigned short* src = qkv_bf + ((size_t)b * 768 + ch) * 16384;
        float hin[36];
#pragma unroll
        for (int iy = 0; iy < 6; ++iy) {
            const int y  = wy * 4 - 1 + iy;
            const int yc = min(max(y, 0), 127);
#pragma unroll
            for (int ix = 0; ix < 6; ++ix) {
                const int x  = wx * 4 - 1 + ix;
                const int xc = min(max(x, 0), 127);
                // unconditional (clamped) load, then 0/1 mask -> no exec branches
                const float inb = (y == yc && x == xc) ? 1.f : 0.f;
                hin[iy * 6 + ix] = bf2f(src[yc * 128 + xc]) * inb;
            }
        }
        float* dst = (s == 0) ? lq : ((s == 1) ? lk : lv);
        const int dstride = (s == 2) ? 16 : 32;
#pragma unroll
        for (int e = 0; e < 16; ++e) {
            const int oy = e >> 2, ox = e & 3;
            float sum = 0.f;
#pragma unroll
            for (int ky = 0; ky < 3; ++ky)
#pragma unroll
                for (int kx = 0; kx < 3; ++kx)
                    sum += w[ky * 3 + kx] * hin[(oy + ky) * 6 + (ox + kx)];
            dst[d * dstride + e] = sum;
        }
    }
    __syncthreads();

    // ---- l2 normalize rows of q and k (lane = row d) ----
    {
        const int d = lane;
        float s1 = 0.f, s2 = 0.f;
#pragma unroll
        for (int e = 0; e < 16; ++e) {
            float a = lq[d * 32 + e]; s1 += a * a;
            float c = lk[d * 32 + e]; s2 += c * c;
        }
        const float r1 = 1.f / fmaxf(sqrtf(s1), 1e-12f);
        const float r2 = 1.f / fmaxf(sqrtf(s2), 1e-12f);
#pragma unroll
        for (int e = 0; e < 16; ++e) { lq[d * 32 + e] *= r1; lk[d * 32 + e] *= r2; }
    }
    __syncthreads();

    const int m    = lane & 15;
    const int half = lane >> 4;

    // ---- logits: attn[d][f] = sum_e qn[d][e]*kn[f][e]; K=16 zero-padded to 32 ----
#pragma unroll
    for (int mt = 0; mt < 2; ++mt)
#pragma unroll
        for (int ntt = 0; ntt < 2; ++ntt) {
            Frag a, bb;
            // A element i -> K = i + 8*half + (i>=8 ? 8 : 0); K>=16 hits zero pad
#pragma unroll
            for (int i = 0; i < 16; ++i) {
                const int e = i + 8 * half + (i >= 8 ? 8 : 0);
                a.v[i] = (__bf16)lq[(mt * 16 + m) * 32 + e];
            }
            // B column f = ntt*16 + m; element i -> K = i + 8*half + (i>=8 ? 8 : 0)
#pragma unroll
            for (int i = 0; i < 16; ++i)
                bb.v[i] = (__bf16)lk[(ntt * 16 + m) * 32 + i + 8 * half + (i >= 8 ? 8 : 0)];
            v8f acc = {0.f, 0.f, 0.f, 0.f, 0.f, 0.f, 0.f, 0.f};
            acc = __builtin_amdgcn_wmma_f32_16x16x32_bf16(
                false, a.v, false, bb.v, (short)0, acc, false, false);
#pragma unroll
            for (int r = 0; r < 8; ++r)
                lat[(mt * 16 + r + 8 * half) * 32 + ntt * 16 + (lane & 15)] = acc[r] * temp;
        }
    __syncthreads();

    // ---- softmax over f (lane = row d) ----
    {
        float* row = lat + lane * 32;
        float mx = row[0];
#pragma unroll
        for (int f = 1; f < 32; ++f) mx = fmaxf(mx, row[f]);
        float s = 0.f;
#pragma unroll
        for (int f = 0; f < 32; ++f) { float e = __expf(row[f] - mx); row[f] = e; s += e; }
        const float inv = 1.f / s;
#pragma unroll
        for (int f = 0; f < 32; ++f) row[f] *= inv;
    }
    __syncthreads();

    // ---- out = attn(32x32) @ v(32x16), K = 32 ----
#pragma unroll
    for (int mt = 0; mt < 2; ++mt) {
        Frag a, bb;
#pragma unroll
        for (int i = 0; i < 16; ++i) {
            const int f = i + 8 * half + (i >= 8 ? 8 : 0);
            a.v[i] = (__bf16)lat[(mt * 16 + m) * 32 + f];
        }
#pragma unroll
        for (int i = 0; i < 16; ++i) bb.v[i] = (__bf16)lv[(i + 8 * half + (i >= 8 ? 8 : 0)) * 16 + m];   // row f = table order, col = m
        v8f acc = {0.f, 0.f, 0.f, 0.f, 0.f, 0.f, 0.f, 0.f};
        acc = __builtin_amdgcn_wmma_f32_16x16x32_bf16(
            false, a.v, false, bb.v, (short)0, acc, false, false);
#pragma unroll
        for (int r = 0; r < 8; ++r) {
            const int d = mt * 16 + r + 8 * half;
            const int e = lane & 15;
            const int y = wy * 4 + (e >> 2);
            const int x = wx * 4 + (e & 3);
            attnout[((size_t)b * 256 + h * 32 + d) * 16384 + y * 128 + x] = f2bf(acc[r]);
        }
    }
}

// ---------- launch ----------
extern "C" void kernel_launch(void* const* d_in, const int* in_sizes, int n_in,
                              void* d_out, int out_size, void* d_ws, size_t ws_size,
                              hipStream_t stream) {
    (void)in_sizes; (void)n_in; (void)out_size; (void)ws_size;
    const float* x      = (const float*)d_in[0];   // (8,256,128,128)
    const float* W_qkv  = (const float*)d_in[1];   // (768,256)
    const float* W_dw   = (const float*)d_in[2];   // (768,1,3,3)
    const float* temper = (const float*)d_in[3];   // (8,1,1)
    const float* W_proj = (const float*)d_in[4];   // (256,256)
    float* out = (float*)d_out;                    // (8,256,128,128) fp32

    const unsigned long long nX  = 33554432ull;    // 8*256*128*128
    const unsigned long long nWq = 768ull * 256ull;
    const unsigned long long nWp = 256ull * 256ull;

    char* ws = (char*)d_ws;
    unsigned short* x_bf    = (unsigned short*)(ws);                       //  64 MB
    unsigned short* Wq_bf   = (unsigned short*)(ws + 67108864ull);         // 384 KB
    unsigned short* Wp_bf   = (unsigned short*)(ws + 67502080ull);         // 128 KB
    unsigned short* qkv_bf  = (unsigned short*)(ws + 67633152ull);         // 192 MB
    unsigned short* att_bf  = (unsigned short*)(ws + 268959744ull);        //  64 MB

    // fp32 -> bf16
    f2bf_kernel<<<(unsigned)((nX  + 255) / 256), 256, 0, stream>>>(x,      x_bf,  nX);
    f2bf_kernel<<<(unsigned)((nWq + 255) / 256), 256, 0, stream>>>(W_qkv,  Wq_bf, nWq);
    f2bf_kernel<<<(unsigned)((nWp + 255) / 256), 256, 0, stream>>>(W_proj, Wp_bf, nWp);

    // qkv = W_qkv @ x  (768 rows, bf16 out)
    gemm_bf16_kernel<1><<<2048, 128, 0, stream>>>(Wq_bf, x_bf, (void*)qkv_bf, 48);

    // fused dwconv + window channel attention (bf16 out in (b,c,h,w) layout)
    attn_kernel<<<16384, 128, 0, stream>>>(qkv_bf, W_dw, temper, att_bf);

    // out = W_proj @ attn_out  (256 rows, fp32 out)
    gemm_bf16_kernel<0><<<2048, 128, 0, stream>>>(Wp_bf, att_bf, (void*)out, 16);
}
